// SparseMoELayer_19189913879366
// MI455X (gfx1250) — matrix operand order repaired and checked by reading, not compiled, not run on hardware
//
#include <hip/hip_runtime.h>
#include <hip/hip_bf16.h>
#include <math.h>

#define BB    8192
#define DD    1024
#define HH    4096
#define DOUTN 1024
#define EE    8
#define KTOP  2

#define LDT   72          // padded LDS row stride (bf16 elems) for a 64-wide K stage

typedef __attribute__((ext_vector_type(16))) __bf16 v16bf;
typedef __attribute__((ext_vector_type(8)))  __bf16 v8bf;
typedef __attribute__((ext_vector_type(8)))  float  v8f;
typedef __attribute__((ext_vector_type(4)))  int    v4i;

#define AS1 __attribute__((address_space(1)))
#define AS3 __attribute__((address_space(3)))

#if defined(__has_builtin)
#if __has_builtin(__builtin_amdgcn_global_load_async_to_lds_b128)
#define HAVE_ASYNC 1
#if __has_builtin(__builtin_amdgcn_s_wait_asynccnt)
#define WAIT_ASYNC() __builtin_amdgcn_s_wait_asynccnt(0)
#else
#define WAIT_ASYNC() asm volatile("s_wait_asynccnt 0x0" ::: "memory")
#endif
#endif
#endif
#ifndef HAVE_ASYNC
#define HAVE_ASYNC 0
#endif

#if HAVE_ASYNC
static __device__ __forceinline__ void async_cp16(const __bf16* g, __bf16* l) {
  __builtin_amdgcn_global_load_async_to_lds_b128((AS1 v4i*)g, (AS3 v4i*)l, 0, 0);
}
#endif

// Build a 16x32 A-frag or 32x16 B-frag lane vector from two contiguous 8-elem chunks.
static __device__ __forceinline__ v16bf frag16(const __bf16* p0, const __bf16* p1) {
  v8bf lo = *(const v8bf*)p0;
  v8bf hi = *(const v8bf*)p1;
  return __builtin_shufflevector(lo, hi, 0,1,2,3,4,5,6,7,8,9,10,11,12,13,14,15);
}

// One K=64 stage of WMMA work from the current LDS buffers.
static __device__ __forceinline__ void compute_stage(const __bf16* Asb, const __bf16* Bsb,
                                                     v8f (&acc)[4][2], int wm, int wn,
                                                     int r16, int akb, int bkb) {
#pragma unroll
  for (int ks = 0; ks < 64; ks += 32) {
    v16bf af[4], bfr[2];
#pragma unroll
    for (int ti = 0; ti < 4; ++ti) {
      const __bf16* p = Asb + (wm * 64 + ti * 16 + r16) * LDT + ks + akb;
      af[ti] = frag16(p, p + 16);
    }
#pragma unroll
    for (int tj = 0; tj < 2; ++tj) {
      const __bf16* p = Bsb + (wn * 32 + tj * 16 + r16) * LDT + ks + bkb;
      bfr[tj] = frag16(p, p + 16);
    }
#pragma unroll
    for (int ti = 0; ti < 4; ++ti)
#pragma unroll
      for (int tj = 0; tj < 2; ++tj)
        acc[ti][tj] = __builtin_amdgcn_wmma_f32_16x16x32_bf16(
            false, af[ti], false, bfr[tj], (short)0, acc[ti][tj], false, false);
  }
}

// ---------------------------------------------------------------- zero / init
__global__ __launch_bounds__(256) void moe_zero(float* __restrict__ out, int n,
                                                int* __restrict__ counts,
                                                float* __restrict__ usage) {
  int i = blockIdx.x * 256 + threadIdx.x;
  if (i < n) out[i] = 0.0f;
  if (i < EE) { counts[i] = 0; usage[i] = 0.0f; }
}

// ---------------------------------------------------------------- x -> bf16
__global__ __launch_bounds__(256) void moe_cvt_x(const float* __restrict__ x,
                                                 __bf16* __restrict__ xbf) {
  int i = (blockIdx.x * 256 + threadIdx.x) * 4;
  float4 v = *(const float4*)(x + i);
  xbf[i + 0] = (__bf16)v.x; xbf[i + 1] = (__bf16)v.y;
  xbf[i + 2] = (__bf16)v.z; xbf[i + 3] = (__bf16)v.w;
}

// ---------------------------------------- per-expert transpose f32[R][C] -> bf16[C][R]
__global__ __launch_bounds__(256) void moe_transpose(const float* __restrict__ in,
                                                     __bf16* __restrict__ out,
                                                     int R, int C) {
  __shared__ float tile[32][33];
  int e = blockIdx.z;
  const float* src = in + (size_t)e * R * C;
  __bf16* dst = out + (size_t)e * R * C;
  int c0 = blockIdx.x * 32, r0 = blockIdx.y * 32;
  int tx = threadIdx.x & 31, ty = threadIdx.x >> 5; // 32 x 8
#pragma unroll
  for (int i = 0; i < 4; ++i) {
    int r = ty + i * 8;
    tile[r][tx] = src[(size_t)(r0 + r) * C + c0 + tx];
  }
  __syncthreads();
#pragma unroll
  for (int i = 0; i < 4; ++i) {
    int r = ty + i * 8; // output row within tile = input column
    dst[(size_t)(c0 + r) * R + r0 + tx] = (__bf16)tile[tx][r];
  }
}

// ---------------------------------------------------------------- gating
__global__ __launch_bounds__(256) void moe_gate(const float* __restrict__ x,
                                                const float* __restrict__ Wg,
                                                const float* __restrict__ bg,
                                                int* __restrict__ counts,
                                                int* __restrict__ tok_list,
                                                float* __restrict__ gate_w,
                                                float* __restrict__ usage) {
  __shared__ float wg_s[DD * EE]; // 32 KB
  __shared__ float bg_s[EE];
  for (int i = threadIdx.x; i < DD * EE; i += 256) wg_s[i] = Wg[i];
  if (threadIdx.x < EE) bg_s[threadIdx.x] = bg[threadIdx.x];
  __syncthreads();

  int wave = threadIdx.x >> 5, lane = threadIdx.x & 31;
  int t = blockIdx.x * 8 + wave;
  const float* xr = x + (size_t)t * DD;
  float acc[EE] = {};
  for (int d = lane; d < DD; d += 32) {
    float xv = xr[d];
    const float* w = &wg_s[d * EE];
#pragma unroll
    for (int e = 0; e < EE; ++e) acc[e] += xv * w[e];
  }
#pragma unroll
  for (int e = 0; e < EE; ++e)
#pragma unroll
    for (int off = 16; off > 0; off >>= 1)
      acc[e] += __shfl_down(acc[e], off, 32);

  if (lane == 0) {
    float l[EE], p[EE];
    float m = -1e30f;
#pragma unroll
    for (int e = 0; e < EE; ++e) { l[e] = acc[e] + bg_s[e]; m = fmaxf(m, l[e]); }
    float s = 0.0f;
#pragma unroll
    for (int e = 0; e < EE; ++e) { p[e] = __expf(l[e] - m); s += p[e]; }
    float inv = 1.0f / s;
    int i0 = 0, i1 = -1; float p0 = -1.0f, p1 = -1.0f;
#pragma unroll
    for (int e = 0; e < EE; ++e) {
      float pe = p[e] * inv;
      atomicAdd(&usage[e], pe);
      if (pe > p0)      { p1 = p0; i1 = i0; p0 = pe; i0 = e; }
      else if (pe > p1) { p1 = pe; i1 = e; }
    }
    float wsum = p0 + p1;
    int pos0 = atomicAdd(&counts[i0], 1);
    tok_list[i0 * BB + pos0] = t;  gate_w[i0 * BB + pos0] = p0 / wsum;
    int pos1 = atomicAdd(&counts[i1], 1);
    tok_list[i1 * BB + pos1] = t;  gate_w[i1 * BB + pos1] = p1 / wsum;
  }
}

// ------------------------------------------------- prefix offsets + aux loss
__global__ void moe_finalize(const int* __restrict__ counts,
                             int* __restrict__ offsets,
                             const float* __restrict__ usage,
                             float* __restrict__ aux_out) {
  if (threadIdx.x == 0) {
    int run = 0;
    const float invE = 1.0f / (float)EE;
    const float logInvE = logf(invE);
    float aux = 0.0f;
    for (int e = 0; e < EE; ++e) {
      offsets[e] = run; run += counts[e];
      float u = usage[e] / (float)BB;
      aux += u * logInvE - logf(u) * invE;
    }
    aux_out[0] = aux;
  }
}

// ---------------------------------------------------------------- GEMM1:
// h[off+m][0..H) = gelu( x[tok[m]] @ W1[e] + b1[e] )   (bf16 wmma, f32 acc)
__global__ __launch_bounds__(256) void moe_gemm1(const __bf16* __restrict__ xbf,
                                                 const __bf16* __restrict__ w1t,
                                                 const float* __restrict__ b1,
                                                 const int* __restrict__ tok_list,
                                                 const int* __restrict__ counts,
                                                 const int* __restrict__ offsets,
                                                 __bf16* __restrict__ hbuf) {
  const int e = blockIdx.z;
  const int cnt = counts[e];
  const int m0 = blockIdx.y * 128;
  if (m0 >= cnt) return;
  const int n0 = blockIdx.x * 128;
  const int off = offsets[e];

  __shared__ __align__(16) __bf16 As[2][128 * LDT];
  __shared__ __align__(16) __bf16 Bs[2][128 * LDT];

  const int tid = threadIdx.x;

  // per-thread staging plan: 4 x 16B chunks each for A and B per K=64 stage
  const __bf16* aSrc[4]; const __bf16* bSrc[4];
  int aDst[4], bDst[4];
#pragma unroll
  for (int i = 0; i < 4; ++i) {
    int idx = tid + i * 256;
    int r = idx >> 3, cc = (idx & 7) * 8;
    int m = m0 + r;
    int tok = tok_list[e * BB + (m < cnt ? m : cnt - 1)];
    aSrc[i] = xbf + (size_t)tok * DD + cc;
    aDst[i] = r * LDT + cc;
    bSrc[i] = w1t + ((size_t)e * HH + n0 + r) * DD + cc;
    bDst[i] = r * LDT + cc;
  }

  const int wave = tid >> 5, lane = tid & 31;
  const int wm = wave & 1, wn = wave >> 1;       // 2 x 4 wave grid
  const int r16 = lane & 15, lh = lane >> 4;
  const int akb = lh * 8, bkb = lh * 8;

  v8f acc[4][2] = {};
  const int iters = DD / 64;

#if HAVE_ASYNC
#pragma unroll
  for (int i = 0; i < 4; ++i) {
    async_cp16(aSrc[i], &As[0][aDst[i]]);
    async_cp16(bSrc[i], &Bs[0][bDst[i]]);
  }
  for (int it = 0; it < iters; ++it) {
    int cur = it & 1;
    WAIT_ASYNC();
    __syncthreads();
    if (it + 1 < iters) {
      int k1 = (it + 1) * 64;
#pragma unroll
      for (int i = 0; i < 4; ++i) {
        async_cp16(aSrc[i] + k1, &As[cur ^ 1][aDst[i]]);
        async_cp16(bSrc[i] + k1, &Bs[cur ^ 1][bDst[i]]);
      }
    }
    compute_stage(As[cur], Bs[cur], acc, wm, wn, r16, akb, bkb);
  }
#else
  v8bf ra[4], rb[4];
#pragma unroll
  for (int i = 0; i < 4; ++i) { ra[i] = *(const v8bf*)aSrc[i]; rb[i] = *(const v8bf*)bSrc[i]; }
#pragma unroll
  for (int i = 0; i < 4; ++i) {
    *(v8bf*)&As[0][aDst[i]] = ra[i];
    *(v8bf*)&Bs[0][bDst[i]] = rb[i];
  }
  for (int it = 0; it < iters; ++it) {
    int cur = it & 1;
    __syncthreads();
    if (it + 1 < iters) {
      int k1 = (it + 1) * 64;
#pragma unroll
      for (int i = 0; i < 4; ++i) {
        ra[i] = *(const v8bf*)(aSrc[i] + k1);
        rb[i] = *(const v8bf*)(bSrc[i] + k1);
      }
    }
    compute_stage(As[cur], Bs[cur], acc, wm, wn, r16, akb, bkb);
    if (it + 1 < iters) {
#pragma unroll
      for (int i = 0; i < 4; ++i) {
        *(v8bf*)&As[cur ^ 1][aDst[i]] = ra[i];
        *(v8bf*)&Bs[cur ^ 1][bDst[i]] = rb[i];
      }
    }
  }
#endif

  // epilogue: + b1, gelu(exact erf), store bf16
  const int mb = lh * 8;
#pragma unroll
  for (int tj = 0; tj < 2; ++tj) {
    int gn = n0 + wn * 32 + tj * 16 + r16;
    float b1v = b1[(size_t)e * HH + gn];
#pragma unroll
    for (int ti = 0; ti < 4; ++ti) {
#pragma unroll
      for (int v = 0; v < 8; ++v) {
        int gm = m0 + wm * 64 + ti * 16 + mb + v;
        if (gm < cnt) {
          float val = acc[ti][tj][v] + b1v;
          float g = 0.5f * val * (1.0f + erff(val * 0.70710678f));
          hbuf[(size_t)(off + gm) * HH + gn] = (__bf16)g;
        }
      }
    }
  }
}

// ---------------------------------------------------------------- GEMM2:
// out[tok[m]] += w[m] * ( h[off+m] @ W2[e] + b2[e] )
__global__ __launch_bounds__(256) void moe_gemm2(const __bf16* __restrict__ hbuf,
                                                 const __bf16* __restrict__ w2t,
                                                 const float* __restrict__ b2,
                                                 const int* __restrict__ tok_list,
                                                 const float* __restrict__ gate_w,
                                                 const int* __restrict__ counts,
                                                 const int* __restrict__ offsets,
                                                 float* __restrict__ out) {
  const int e = blockIdx.z;
  const int cnt = counts[e];
  const int m0 = blockIdx.y * 128;
  if (m0 >= cnt) return;
  const int n0 = blockIdx.x * 128;
  const int off = offsets[e];

  __shared__ __align__(16) __bf16 As[2][128 * LDT];
  __shared__ __align__(16) __bf16 Bs[2][128 * LDT];
  __shared__ int stok[128];
  __shared__ float gws[128];

  const int tid = threadIdx.x;
  if (tid < 128) {
    int m = m0 + tid;
    int mc = m < cnt ? m : cnt - 1;
    stok[tid] = tok_list[e * BB + mc];
    gws[tid] = gate_w[e * BB + mc];
  }

  const __bf16* aSrc[4]; const __bf16* bSrc[4];
  int aDst[4], bDst[4];
#pragma unroll
  for (int i = 0; i < 4; ++i) {
    int idx = tid + i * 256;
    int r = idx >> 3, cc = (idx & 7) * 8;
    int m = m0 + r;
    int arow = off + (m < cnt ? m : cnt - 1);
    aSrc[i] = hbuf + (size_t)arow * HH + cc;
    aDst[i] = r * LDT + cc;
    bSrc[i] = w2t + ((size_t)e * DOUTN + n0 + r) * HH + cc;
    bDst[i] = r * LDT + cc;
  }

  const int wave = tid >> 5, lane = tid & 31;
  const int wm = wave & 1, wn = wave >> 1;
  const int r16 = lane & 15, lh = lane >> 4;
  const int akb = lh * 8, bkb = lh * 8;

  v8f acc[4][2] = {};
  const int iters = HH / 64;

#if HAVE_ASYNC
#pragma unroll
  for (int i = 0; i < 4; ++i) {
    async_cp16(aSrc[i], &As[0][aDst[i]]);
    async_cp16(bSrc[i], &Bs[0][bDst[i]]);
  }
  for (int it = 0; it < iters; ++it) {
    int cur = it & 1;
    WAIT_ASYNC();
    __syncthreads();
    if (it + 1 < iters) {
      int k1 = (it + 1) * 64;
#pragma unroll
      for (int i = 0; i < 4; ++i) {
        async_cp16(aSrc[i] + k1, &As[cur ^ 1][aDst[i]]);
        async_cp16(bSrc[i] + k1, &Bs[cur ^ 1][bDst[i]]);
      }
    }
    compute_stage(As[cur], Bs[cur], acc, wm, wn, r16, akb, bkb);
  }
#else
  v8bf ra[4], rb[4];
#pragma unroll
  for (int i = 0; i < 4; ++i) { ra[i] = *(const v8bf*)aSrc[i]; rb[i] = *(const v8bf*)bSrc[i]; }
#pragma unroll
  for (int i = 0; i < 4; ++i) {
    *(v8bf*)&As[0][aDst[i]] = ra[i];
    *(v8bf*)&Bs[0][bDst[i]] = rb[i];
  }
  for (int it = 0; it < iters; ++it) {
    int cur = it & 1;
    __syncthreads();
    if (it + 1 < iters) {
      int k1 = (it + 1) * 64;
#pragma unroll
      for (int i = 0; i < 4; ++i) {
        ra[i] = *(const v8bf*)(aSrc[i] + k1);
        rb[i] = *(const v8bf*)(bSrc[i] + k1);
      }
    }
    compute_stage(As[cur], Bs[cur], acc, wm, wn, r16, akb, bkb);
    if (it + 1 < iters) {
#pragma unroll
      for (int i = 0; i < 4; ++i) {
        *(v8bf*)&As[cur ^ 1][aDst[i]] = ra[i];
        *(v8bf*)&Bs[cur ^ 1][bDst[i]] = rb[i];
      }
    }
  }
#endif

  const int mb = lh * 8;
#pragma unroll
  for (int tj = 0; tj < 2; ++tj) {
    int gn = n0 + wn * 32 + tj * 16 + r16;
    float b2v = b2[(size_t)e * DOUTN + gn];
#pragma unroll
    for (int ti = 0; ti < 4; ++ti) {
#pragma unroll
      for (int v = 0; v < 8; ++v) {
        int gm = m0 + wm * 64 + ti * 16 + mb + v;
        if (gm < cnt) {
          int li = gm - m0;
          float y = (acc[ti][tj][v] + b2v) * gws[li];
          atomicAdd(&out[(size_t)stok[li] * DOUTN + gn], y);
        }
      }
    }
  }
}

// ---------------------------------------------------------------- launcher
extern "C" void kernel_launch(void* const* d_in, const int* in_sizes, int n_in,
                              void* d_out, int out_size, void* d_ws, size_t ws_size,
                              hipStream_t stream) {
  const float* x  = (const float*)d_in[0];
  const float* Wg = (const float*)d_in[1];
  const float* bg = (const float*)d_in[2];
  const float* W1 = (const float*)d_in[3];
  const float* b1 = (const float*)d_in[4];
  const float* W2 = (const float*)d_in[5];
  const float* b2 = (const float*)d_in[6];
  float* out = (float*)d_out;

  char* ws = (char*)d_ws;
  size_t o = 0;
  __bf16* xbf  = (__bf16*)(ws + o); o += (size_t)BB * DD * 2;
  __bf16* w1t  = (__bf16*)(ws + o); o += (size_t)EE * HH * DD * 2;
  __bf16* w2t  = (__bf16*)(ws + o); o += (size_t)EE * DOUTN * HH * 2;
  __bf16* hbuf = (__bf16*)(ws + o); o += (size_t)BB * KTOP * HH * 2;
  int*   tok   = (int*)  (ws + o); o += (size_t)EE * BB * 4;
  float* gw    = (float*)(ws + o); o += (size_t)EE * BB * 4;
  int*   counts  = (int*)  (ws + o); o += 256;
  int*   offsets = (int*)  (ws + o); o += 256;
  float* usage   = (float*)(ws + o); o += 256;

  // 1. zero output accumulators + counters
  moe_zero<<<(BB * DOUTN + 255) / 256, 256, 0, stream>>>(out, BB * DOUTN, counts, usage);

  // 2. x -> bf16
  moe_cvt_x<<<(BB * DD / 4) / 256, 256, 0, stream>>>(x, xbf);

  // 3. W1 [E][D][H] -> bf16 [E][H][D]; W2 [E][H][DOUT] -> bf16 [E][DOUT][H]
  moe_transpose<<<dim3(HH / 32, DD / 32, EE), 256, 0, stream>>>(W1, w1t, DD, HH);
  moe_transpose<<<dim3(DOUTN / 32, HH / 32, EE), 256, 0, stream>>>(W2, w2t, HH, DOUTN);

  // 4. gating + routing
  moe_gate<<<BB / 8, 256, 0, stream>>>(x, Wg, bg, counts, tok, gw, usage);

  // 5. prefix offsets + aux loss
  moe_finalize<<<1, 32, 0, stream>>>(counts, offsets, usage, out + (size_t)BB * DOUTN);

  // 6. expert GEMM1 (gelu) -> hbuf
  moe_gemm1<<<dim3(HH / 128, BB / 128, EE), 256, 0, stream>>>(
      xbf, w1t, b1, tok, counts, offsets, hbuf);

  // 7. expert GEMM2 -> weighted atomic combine into out
  moe_gemm2<<<dim3(DOUTN / 128, BB / 128, EE), 256, 0, stream>>>(
      hbuf, w2t, b2, tok, gw, counts, offsets, out);
}
